// Decoder_3161095930296
// MI455X (gfx1250) — matrix operand order repaired and checked by reading, not compiled, not run on hardware
//
#include <hip/hip_runtime.h>
#include <hip/hip_bf16.h>

// Problem constants (from reference)
#define B_  64
#define L_  2048
#define H_  256
#define E_  256
#define V_  50257
#define TWOH 512
#define XW  768          // E + 2H
#define ROWSTRIDE (V_ + L_)

typedef __bf16 v16bf __attribute__((ext_vector_type(16)));
typedef float  v8f   __attribute__((ext_vector_type(8)));

// fp32 -> bf16 (round to nearest even), raw bits
__device__ __forceinline__ unsigned short f2bf(float f) {
    unsigned int u = __float_as_uint(f);
    u += 0x7FFFu + ((u >> 16) & 1u);
    return (unsigned short)(u >> 16);
}
__device__ __forceinline__ unsigned int pack2bf(float lo, float hi) {
    return (unsigned int)f2bf(lo) | ((unsigned int)f2bf(hi) << 16);
}

// ---------------------------------------------------------------------------
// K0: pre-swizzle Wc (512x256 fp32) -> bf16 in WMMA B-fragment layout:
//   dst[((ks*16 + nt)*512) + ((n&15) + 16*((k&31)>>4))*16 + (k&15 of kl)]
// so a lane's v16bf fragment is 32 contiguous bytes. grid = 512 x 256.
// ---------------------------------------------------------------------------
__global__ __launch_bounds__(256) void k0_swizzle_wc(
    const float* __restrict__ WcW, unsigned short* __restrict__ wc_sw)
{
    const int idx = blockIdx.x * 256 + threadIdx.x;   // 0..131071
    const int k = idx >> 8, n = idx & 255;
    const float f = WcW[idx];
    const int ks = k >> 5, kl = k & 31;
    const int nt = n >> 4, nl = n & 15;
    wc_sw[(size_t)(ks * 16 + nt) * 512 + (nl + 16 * (kl >> 4)) * 16 + (kl & 15)] = f2bf(f);
}

// ---------------------------------------------------------------------------
// K1: embedding gather + attention logits; also writes emb into x[:,0:256]
// and zeroes x ctx columns. grid = B*8 blocks of 256.
// ---------------------------------------------------------------------------
__global__ __launch_bounds__(256) void k1_embed_attn(
    const int* __restrict__ input_idx, const float* __restrict__ embed_table,
    const float* __restrict__ pre_state, const float* __restrict__ attn_W,
    const float* __restrict__ attn_b, float* __restrict__ x,
    float* __restrict__ logits)
{
    const int b  = blockIdx.x >> 3;
    const int lc = blockIdx.x & 7;
    const int tid = threadIdx.x;
    __shared__ float s_in[TWOH];

    const int row = input_idx[b];
    const float e0 = embed_table[(size_t)row * E_ + tid];
    s_in[tid]      = e0;
    s_in[E_ + tid] = pre_state[b * H_ + tid];
    if (lc == 0) x[b * XW + tid] = e0;              // x left block = emb
    if (lc == 1) x[b * XW + E_ + tid] = 0.f;        // zero ctx (atomics later)
    if (lc == 2) x[b * XW + E_ + 256 + tid] = 0.f;
    __syncthreads();

    const int l = lc * 256 + tid;
    float acc = attn_b[l];
    #pragma unroll 4
    for (int k = 0; k < TWOH; ++k)
        acc = fmaf(s_in[k], attn_W[(size_t)k * L_ + l], acc);
    logits[b * L_ + l] = acc;
}

// ---------------------------------------------------------------------------
// K2: per-row softmax + fused copy-select weights. grid = B blocks.
// ---------------------------------------------------------------------------
__global__ __launch_bounds__(256) void k2_combine(
    const int* __restrict__ input_idx, const int* __restrict__ seq,
    const float* __restrict__ pre_prob_c, float* __restrict__ w)
{
    const int b = blockIdx.x, tid = threadIdx.x;
    __shared__ float red[256];
    __shared__ int  redi[256];
    const int myidx = input_idx[b];

    float m = -3.402823466e+38f;
    int cnt = 0;
    for (int l = tid; l < L_; l += 256) {
        m = fmaxf(m, w[b * L_ + l]);
        cnt += (seq[b * L_ + l] == myidx) ? 1 : 0;
    }
    red[tid] = m; redi[tid] = cnt; __syncthreads();
    for (int s = 128; s > 0; s >>= 1) {
        if (tid < s) { red[tid] = fmaxf(red[tid], red[tid + s]); redi[tid] += redi[tid + s]; }
        __syncthreads();
    }
    m = red[0]; cnt = redi[0]; __syncthreads();

    float sum = 0.f;
    for (int l = tid; l < L_; l += 256) sum += expf(w[b * L_ + l] - m);
    red[tid] = sum; __syncthreads();
    for (int s = 128; s > 0; s >>= 1) {
        if (tid < s) red[tid] += red[tid + s];
        __syncthreads();
    }
    const float inv  = 1.f / red[0];
    const float rcnt = 1.f / fmaxf((float)cnt, 1.f);
    for (int l = tid; l < L_; l += 256) {
        float aw  = expf(w[b * L_ + l] - m) * inv;
        float sel = (seq[b * L_ + l] == myidx) ? pre_prob_c[b * L_ + l] * rcnt : 0.f;
        w[b * L_ + l] = aw + sel;
    }
}

// ---------------------------------------------------------------------------
// K3: ctx[b,d] = sum_l w[b,l]*enc[b,l,d]  (enc pass 1, pure bandwidth)
// grid = B*2*4 = 512 blocks; atomicAdd partials into x ctx.
// ---------------------------------------------------------------------------
__global__ __launch_bounds__(256) void k3_context(
    const float* __restrict__ enc, const float* __restrict__ cw,
    float* __restrict__ x)
{
    const int blk  = blockIdx.x;
    const int b    = blk >> 3;
    const int half = (blk >> 2) & 1;
    const int lq   = blk & 3;
    const int tid  = threadIdx.x;
    const int d    = half * 256 + tid;
    __shared__ float s_w[256];

    const float* eb = enc + (size_t)b * L_ * TWOH;
    const int lbase = lq * 512;
    float acc = 0.f;
    for (int l0 = 0; l0 < 512; l0 += 256) {
        __syncthreads();
        s_w[tid] = cw[b * L_ + lbase + l0 + tid];
        __syncthreads();
        #pragma unroll 4
        for (int l = 0; l < 256; ++l)
            acc = fmaf(s_w[l], eb[(size_t)(lbase + l0 + l) * TWOH + d], acc);
    }
    atomicAdd(&x[b * XW + E_ + d], acc);
}

// ---------------------------------------------------------------------------
// K4: GRU cell -> state (written directly to state output slot). grid = B.
// ---------------------------------------------------------------------------
__global__ __launch_bounds__(256) void k4_gru(
    const float* __restrict__ x, const float* __restrict__ pre_state,
    const float* __restrict__ Wih, const float* __restrict__ Whh,
    const float* __restrict__ bih, const float* __restrict__ bhh,
    float* __restrict__ state)
{
    const int b = blockIdx.x, tid = threadIdx.x;
    __shared__ float s_x[XW], s_h[H_], s_gi[3 * H_], s_gh[3 * H_];
    for (int i = tid; i < XW; i += 256) s_x[i] = x[b * XW + i];
    s_h[tid] = pre_state[b * H_ + tid];
    __syncthreads();

    for (int j = tid; j < 3 * H_; j += 256) {
        float a1 = bih[j], a2 = bhh[j];
        const float* wi = Wih + (size_t)j * XW;
        const float* wh = Whh + (size_t)j * H_;
        #pragma unroll 4
        for (int k = 0; k < XW; ++k) a1 = fmaf(s_x[k], wi[k], a1);
        #pragma unroll 4
        for (int k = 0; k < H_; ++k) a2 = fmaf(s_h[k], wh[k], a2);
        s_gi[j] = a1; s_gh[j] = a2;
    }
    __syncthreads();

    const float r = 1.f / (1.f + expf(-(s_gi[tid] + s_gh[tid])));
    const float z = 1.f / (1.f + expf(-(s_gi[H_ + tid] + s_gh[H_ + tid])));
    const float n = tanhf(s_gi[2 * H_ + tid] + r * s_gh[2 * H_ + tid]);
    state[b * H_ + tid] = (1.f - z) * n + z * s_h[tid];
}

// ---------------------------------------------------------------------------
// K5: score_g = state @ Wo + Wo_b via v_wmma_f32_16x16x32_bf16
// M=64, K=256, N-tile=32/block. grid = ceil(V/32). Branchless OOB handling.
// ---------------------------------------------------------------------------
__global__ __launch_bounds__(256) void k5_score_g(
    const float* __restrict__ state, const float* __restrict__ WoW,
    const float* __restrict__ Wob, float* __restrict__ out)
{
    const int n0  = blockIdx.x * 32;
    const int tid = threadIdx.x;
    const int lane = tid & 31, wave = tid >> 5;
    __shared__ __align__(32) unsigned short Abuf[4 * 8 * 512]; // 4 m-tiles x 8 k-steps
    __shared__ __align__(32) unsigned short Bbuf[2 * 512];     // 2 n-tiles

    // stage A = state (64x256) bf16 swizzled; 8 K per thread-chunk -> 16B LDS store
    #pragma unroll
    for (int e = 0; e < 8; ++e) {
        const int c  = tid + e * 256;      // chunk 0..2047
        const int m  = c >> 5;             // 32 chunks (of 8 K) per row
        const int kc = (c & 31) * 8;
        const float4 f0 = *(const float4*)(state + m * H_ + kc);
        const float4 f1 = *(const float4*)(state + m * H_ + kc + 4);
        uint4 p;
        p.x = pack2bf(f0.x, f0.y); p.y = pack2bf(f0.z, f0.w);
        p.z = pack2bf(f1.x, f1.y); p.w = pack2bf(f1.z, f1.w);
        const int ks = kc >> 5, kl = kc & 31;
        const int lx = (m & 15) + 16 * (kl >> 4);
        const int i0 = 8 * ((kl >> 3) & 1);
        *(uint4*)(&Abuf[((m >> 4) * 8 + ks) * 512 + lx * 16 + i0]) = p;
    }

    const int mt  = wave & 3;
    const int ntl = wave >> 2;
    v8f acc = {};

    for (int ks = 0; ks < 8; ++ks) {
        __syncthreads();
        // stage B block Wo[ks*32..+32][n0..n0+32), branchless zero-pad past V
        #pragma unroll
        for (int e = 0; e < 4; ++e) {
            const int idx = tid + e * 256;
            const int k = idx >> 5, n = idx & 31;
            const int gn = n0 + n;
            const int gnc = (gn < V_) ? gn : (V_ - 1);
            float f = WoW[(size_t)(ks * 32 + k) * V_ + gnc];
            f = (gn < V_) ? f : 0.f;
            const int nt = n >> 4, nl = n & 15;
            Bbuf[nt * 512 + (nl + 16 * (k >> 4)) * 16 + (k & 15)] = f2bf(f);
        }
        __syncthreads();
        const v16bf a  = *(const v16bf*)(&Abuf[(mt * 8 + ks) * 512 + lane * 16]);
        const v16bf bb = *(const v16bf*)(&Bbuf[ntl * 512 + lane * 16]);
        acc = __builtin_amdgcn_wmma_f32_16x16x32_bf16(false, a, false, bb,
                                                      (short)0, acc, false, false);
    }

    const int gn = n0 + ntl * 16 + (lane & 15);
    if (gn < V_) {
        const float bias = Wob[gn];
        const int rbase = mt * 16 + (lane >> 4) * 8;
        #pragma unroll
        for (int r = 0; r < 8; ++r)
            out[(size_t)(rbase + r) * ROWSTRIDE + gn] = acc[r] + bias;
    }
}

// ---------------------------------------------------------------------------
// K6: score_c[b,l] = tanh(enc[b,l,:] @ Wc + Wc_b) . state[b,:]
// Dominant GEMM M=131072,K=512,N=256. Block = 64 enc rows; 8 waves, each owning
// 2 m-tiles x 4 n-tiles (B fragment reused for 2 wmma). A staged through LDS
// (fp32 HBM stream -> bf16 packed b128 stores); B fragments loaded directly
// from the pre-swizzled global copy (L2-resident, fully coalesced b128 loads).
// grid = B*L/64 = 2048 blocks.
// ---------------------------------------------------------------------------
__global__ __launch_bounds__(256) void k6_score_c(
    const float* __restrict__ enc, const unsigned short* __restrict__ wc_sw,
    const float* __restrict__ Wcb, const float* __restrict__ state,
    float* __restrict__ score_c)
{
    const int rows0 = blockIdx.x * 64;       // 64 rows, same b (2048 % 64 == 0)
    const int b = rows0 >> 11;
    const int tid = threadIdx.x;
    const int lane = tid & 31, wave = tid >> 5;

    __shared__ __align__(32) unsigned short Abuf[4 * 512];   // 64 rows x 32 K
    __shared__ float s_state[H_];
    __shared__ float s_bias[H_];
    __shared__ float s_sc[64];

    if (tid < 64) s_sc[tid] = 0.f;
    s_state[tid] = state[b * H_ + tid];
    s_bias[tid]  = Wcb[tid];

    const int mp = wave & 1;      // m-pair: m-tiles {2*mp, 2*mp+1}
    const int q  = wave >> 1;     // n-quad: n-tiles q*4 .. q*4+3
    v8f acc[2][4] = {};

    for (int ks = 0; ks < 16; ++ks) {
        const int kk = ks * 32;
        __syncthreads();
        // stage A: each thread: one 8-K chunk of one row -> 2 x b128 global
        // loads (fp32), RNE-pack, single b128 LDS store (swizzle-contiguous)
        {
            const int m  = tid >> 2;          // 4 chunks per 32-K row slice
            const int kc = (tid & 3) * 8;
            const float* src = enc + (size_t)(rows0 + m) * TWOH + kk + kc;
            const float4 f0 = *(const float4*)(src);
            const float4 f1 = *(const float4*)(src + 4);
            uint4 p;
            p.x = pack2bf(f0.x, f0.y); p.y = pack2bf(f0.z, f0.w);
            p.z = pack2bf(f1.x, f1.y); p.w = pack2bf(f1.z, f1.w);
            const int lx = (m & 15) + 16 * (kc >> 4);
            const int i0 = 8 * ((kc >> 3) & 1);
            *(uint4*)(&Abuf[(m >> 4) * 512 + lx * 16 + i0]) = p;
        }
        __syncthreads();
        const v16bf a0 = *(const v16bf*)(&Abuf[(mp * 2 + 0) * 512 + lane * 16]);
        const v16bf a1 = *(const v16bf*)(&Abuf[(mp * 2 + 1) * 512 + lane * 16]);
        #pragma unroll
        for (int j = 0; j < 4; ++j) {
            const v16bf bb = *(const v16bf*)(
                wc_sw + (size_t)(ks * 16 + q * 4 + j) * 512 + lane * 16);
            acc[0][j] = __builtin_amdgcn_wmma_f32_16x16x32_bf16(false, a0, false, bb,
                                                                (short)0, acc[0][j], false, false);
            acc[1][j] = __builtin_amdgcn_wmma_f32_16x16x32_bf16(false, a1, false, bb,
                                                                (short)0, acc[1][j], false, false);
        }
    }
    __syncthreads();

    // epilogue: tanh(+bias), dot with state, per-row reduce via LDS atomics
    const int nl = lane & 15;
    const int rh = lane >> 4;
    #pragma unroll
    for (int p = 0; p < 2; ++p) {
        const int mt = mp * 2 + p;
        float part[8];
        #pragma unroll
        for (int r = 0; r < 8; ++r) part[r] = 0.f;
        #pragma unroll
        for (int j = 0; j < 4; ++j) {
            const int n = (q * 4 + j) * 16 + nl;
            const float bias = s_bias[n];
            const float sv   = s_state[n];
            #pragma unroll
            for (int r = 0; r < 8; ++r)
                part[r] += tanhf(acc[p][j][r] + bias) * sv;
        }
        #pragma unroll
        for (int r = 0; r < 8; ++r)
            atomicAdd(&s_sc[mt * 16 + rh * 8 + r], part[r]);
    }
    __syncthreads();
    if (tid < 64) score_c[rows0 + tid] = s_sc[tid];
}

// ---------------------------------------------------------------------------
// K7: joint softmax over [score_g | score_c] per row, in place; prob_c scatter
// into the out row (global_atomic_add_f32); zero out[:, V:V+L]. grid = B.
// ---------------------------------------------------------------------------
__global__ __launch_bounds__(256) void k7_final(
    const int* __restrict__ seq, float* __restrict__ out, float* __restrict__ pc)
{
    const int b = blockIdx.x, tid = threadIdx.x;
    __shared__ float red[256];
    float* orow = out + (size_t)b * ROWSTRIDE;
    float* crow = pc + (size_t)b * L_;

    for (int i = V_ + tid; i < ROWSTRIDE; i += 256) orow[i] = 0.f;

    float m = -3.402823466e+38f;
    for (int i = tid; i < V_; i += 256) m = fmaxf(m, orow[i]);
    for (int i = tid; i < L_; i += 256) m = fmaxf(m, crow[i]);
    red[tid] = m; __syncthreads();
    for (int s = 128; s > 0; s >>= 1) {
        if (tid < s) red[tid] = fmaxf(red[tid], red[tid + s]);
        __syncthreads();
    }
    m = red[0]; __syncthreads();

    float sum = 0.f;
    for (int i = tid; i < V_; i += 256) sum += expf(orow[i] - m);
    for (int i = tid; i < L_; i += 256) sum += expf(crow[i] - m);
    red[tid] = sum; __syncthreads();
    for (int s = 128; s > 0; s >>= 1) {
        if (tid < s) red[tid] += red[tid + s];
        __syncthreads();
    }
    const float inv = 1.f / red[0];

    for (int i = tid; i < V_; i += 256) orow[i] = expf(orow[i] - m) * inv;
    __syncthreads();   // prob_g fully written before same-row scatter adds

    const int* srow = seq + b * L_;
    for (int i = tid; i < L_; i += 256) {
        const float p = expf(crow[i] - m) * inv;
        crow[i] = p;                       // prob_c output (in place)
        atomicAdd(&orow[srow[i]], p);      // index_add scatter (same row only)
    }
}

// ---------------------------------------------------------------------------
extern "C" void kernel_launch(void* const* d_in, const int* in_sizes, int n_in,
                              void* d_out, int out_size, void* d_ws, size_t ws_size,
                              hipStream_t stream) {
    const int*   input_idx   = (const int*)  d_in[0];
    const float* enc         = (const float*)d_in[1];
    const int*   seq         = (const int*)  d_in[2];
    const float* pre_state   = (const float*)d_in[3];
    const float* pre_prob_c  = (const float*)d_in[4];
    const float* embed_table = (const float*)d_in[5];
    const float* attn_W      = (const float*)d_in[6];
    const float* attn_b      = (const float*)d_in[7];
    const float* gru_Wih     = (const float*)d_in[8];
    const float* gru_Whh     = (const float*)d_in[9];
    const float* gru_bih     = (const float*)d_in[10];
    const float* gru_bhh     = (const float*)d_in[11];
    const float* Wc_W        = (const float*)d_in[12];
    const float* Wc_b        = (const float*)d_in[13];
    const float* Wo_W        = (const float*)d_in[14];
    const float* Wo_b        = (const float*)d_in[15];

    float* out       = (float*)d_out;                      // B x (V+L)
    float* state_out = out + (size_t)B_ * ROWSTRIDE;       // B x H
    float* prob_c    = state_out + (size_t)B_ * H_;        // B x L (score_c first)

    float* ws_x  = (float*)d_ws;                           // B x 768 = [emb | ctx]
    float* ws_cw = ws_x + (size_t)B_ * XW;                 // B x L (logits->weights)
    unsigned short* ws_wc = (unsigned short*)(ws_cw + (size_t)B_ * L_); // Wc bf16 swizzled

    k0_swizzle_wc<<<512, 256, 0, stream>>>(Wc_W, ws_wc);
    k1_embed_attn<<<B_ * 8, 256, 0, stream>>>(input_idx, embed_table, pre_state,
                                              attn_W, attn_b, ws_x, ws_cw);
    k2_combine<<<B_, 256, 0, stream>>>(input_idx, seq, pre_prob_c, ws_cw);
    k3_context<<<B_ * 8, 256, 0, stream>>>(enc, ws_cw, ws_x);
    k4_gru<<<B_, 256, 0, stream>>>(ws_x, pre_state, gru_Wih, gru_Whh,
                                   gru_bih, gru_bhh, state_out);
    k5_score_g<<<(V_ + 31) / 32, 256, 0, stream>>>(state_out, Wo_W, Wo_b, out);
    k6_score_c<<<(B_ * L_) / 64, 256, 0, stream>>>(enc, ws_wc, Wc_b, state_out, prob_c);
    k7_final<<<B_, 256, 0, stream>>>(seq, out, prob_c);
}
